// SwinTransformerBlock_1975684956945
// MI455X (gfx1250) — matrix operand order repaired and checked by reading, not compiled, not run on hardware
//
#include <hip/hip_runtime.h>
#include <hip/hip_bf16.h>
#include <math.h>

// ---------------------------------------------------------------------------
// Swin transformer block for MI455X (gfx1250, wave32, WMMA bf16 -> f32 accum)
// GEMMs: 32x64 tile per wave (8 WMMAs / k-step, 1.5 b128 loads per WMMA)
// ---------------------------------------------------------------------------

typedef __attribute__((ext_vector_type(8)))  float  v8f;
typedef __attribute__((ext_vector_type(16))) __bf16 v16bf;
typedef __attribute__((ext_vector_type(8)))  __bf16 v8bf;
typedef __attribute__((ext_vector_type(4)))  __bf16 v4bf;

#define NTOK   131072   // 8 * 256 windows * 64 tokens
#define CDIM   512
#define HIDDEN 2048

// ---- workspace layout (bytes) ----
static constexpr size_t OFF_WQT = 0;                                   // 512x512 bf16
static constexpr size_t OFF_WKT = OFF_WQT + (size_t)512 * 512 * 2;
static constexpr size_t OFF_WVT = OFF_WKT + (size_t)512 * 512 * 2;
static constexpr size_t OFF_W1T = OFF_WVT + (size_t)512 * 512 * 2;     // 2048x512 bf16
static constexpr size_t OFF_W2T = OFF_W1T + (size_t)2048 * 512 * 2;    // 512x2048 bf16
static constexpr size_t OFF_H   = 16u * 1024 * 1024;                   // [NTOK,512] bf16
static constexpr size_t SZ_ACT  = (size_t)NTOK * CDIM * 2;             // 128 MiB
static constexpr size_t OFF_Q   = OFF_H + SZ_ACT;
static constexpr size_t OFF_K   = OFF_Q + SZ_ACT;
static constexpr size_t OFF_V   = OFF_K + SZ_ACT;                      // V stored transposed [d,tok]
static constexpr size_t OFF_O   = OFF_V + SZ_ACT;
static constexpr size_t OFF_ATT = OFF_O + SZ_ACT;                      // f32 [NTOK,512]
static constexpr size_t OFF_FIN = OFF_ATT + (size_t)NTOK * CDIM * 4;   // bf16 [NTOK,512]
static constexpr size_t OFF_HID = OFF_H;  // reuse H/Q/K/V (dead after attention): 512 MiB

// ---- helpers --------------------------------------------------------------
__device__ __forceinline__ __bf16 f2bf(float f) {
  union { float f; unsigned u; } x; x.f = f;
  unsigned r = x.u + 0x7fffu + ((x.u >> 16) & 1u);   // round-to-nearest-even
  unsigned short h = (unsigned short)(r >> 16);
  return __builtin_bit_cast(__bf16, h);
}
__device__ __forceinline__ float bf2f(__bf16 b) {
  unsigned short h = __builtin_bit_cast(unsigned short, b);
  union { unsigned u; float f; } x; x.u = ((unsigned)h) << 16;
  return x.f;
}

// A-fragment (16x32 bf16): lane holds row L%16, elems 0..15 = K k0+(L/16)*16 .. +15,
// kb = k0 + (L/16)*8 -> two contiguous 16B loads.
__device__ __forceinline__ v16bf load_a_frag(const __bf16* arow, int kb) {
  kb += kb & 8;
  v8bf lo = *(const v8bf*)(arow + kb);
  v8bf hi = *(const v8bf*)(arow + kb + 8);
  return __builtin_shufflevector(lo, hi, 0, 1, 2, 3, 4, 5, 6, 7,
                                         8, 9, 10, 11, 12, 13, 14, 15);
}

__device__ __forceinline__ v8f wmma_bf16(v16bf a, v16bf b, v8f c) {
  return __builtin_amdgcn_wmma_f32_16x16x32_bf16(false, a, false, b,
                                                 (short)0, c, false, false);
}

// token -> rolled window coordinates
__device__ __forceinline__ void decode_token(int t, int& b, int& win, int& tok) {
  b = t >> 14; win = (t >> 6) & 255; tok = t & 63;
}
__device__ __forceinline__ void token_to_hw(int win, int tok, int& hh, int& ww) {
  int gy = win >> 4, gx = win & 15;
  int py = tok >> 3, px = tok & 7;
  hh = py * 16 + gy;           // rolled-domain coordinates (dilated windows)
  ww = px * 16 + gx;
}

// ---------------------------------------------------------------------------
// K0: weight transpose + bf16 convert: wt[n*K + k] = w[k*N + n]
// ---------------------------------------------------------------------------
__global__ __launch_bounds__(256) void k0_transpose(const float* __restrict__ w,
                                                    __bf16* __restrict__ wt,
                                                    int K, int N) {
  int idx = blockIdx.x * 256 + threadIdx.x;
  if (idx >= K * N) return;
  int n = idx % N, k = idx / N;
  wt[(size_t)n * K + k] = f2bf(w[idx]);
}

// ---------------------------------------------------------------------------
// K1: roll + window gather + LayerNorm1 -> bf16 H[t, 512].  One wave / token.
// ---------------------------------------------------------------------------
__global__ __launch_bounds__(256) void k1_ln1_gather(const float* __restrict__ x,
                                                     const float* __restrict__ g,
                                                     const float* __restrict__ beta,
                                                     __bf16* __restrict__ H) {
  int t = (blockIdx.x * 256 + threadIdx.x) >> 5;
  int lane = threadIdx.x & 31;
  int b, win, tok; decode_token(t, b, win, tok);
  int hh, ww; token_to_hw(win, tok, hh, ww);
  int sh = (hh + 4) & 127, sw = (ww + 4) & 127;   // roll(-4,-4) gather
  const float* src = x + (((size_t)b * 128 + sh) * 128 + sw) * CDIM;
  float4 vb[4];
  float s = 0.f, s2 = 0.f;
#pragma unroll
  for (int j = 0; j < 4; ++j) {
    float4 v = *(const float4*)(src + (j * 32 + lane) * 4);
    vb[j] = v;
    s  += v.x + v.y + v.z + v.w;
    s2 += v.x * v.x + v.y * v.y + v.z * v.z + v.w * v.w;
  }
#pragma unroll
  for (int off = 16; off > 0; off >>= 1) {
    s  += __shfl_xor(s, off, 32);
    s2 += __shfl_xor(s2, off, 32);
  }
  float mu = s * (1.f / CDIM);
  float rstd = rsqrtf(s2 * (1.f / CDIM) - mu * mu + 1e-5f);
  __bf16* dst = H + (size_t)t * CDIM;
#pragma unroll
  for (int j = 0; j < 4; ++j) {
    int c0 = (j * 32 + lane) * 4;
    float4 v = vb[j];
    v4bf o;
    o[0] = f2bf((v.x - mu) * rstd * g[c0 + 0] + beta[c0 + 0]);
    o[1] = f2bf((v.y - mu) * rstd * g[c0 + 1] + beta[c0 + 1]);
    o[2] = f2bf((v.z - mu) * rstd * g[c0 + 2] + beta[c0 + 2]);
    o[3] = f2bf((v.w - mu) * rstd * g[c0 + 3] + beta[c0 + 3]);
    *(v4bf*)(dst + c0) = o;
  }
}

// ---------------------------------------------------------------------------
// K2: QKV projection.  Wave computes 32 rows x 64 cols; z = {q,k,v}.
// Q,K stored [b,h,win,tok,d]; V stored transposed [b,h,win,d,tok].
// ---------------------------------------------------------------------------
__global__ __launch_bounds__(256) void k2_qkv(const __bf16* __restrict__ H,
                                              const __bf16* __restrict__ wqt,
                                              const __bf16* __restrict__ wkt,
                                              const __bf16* __restrict__ wvt,
                                              const float* __restrict__ bq,
                                              const float* __restrict__ bk,
                                              const float* __restrict__ bv,
                                              __bf16* __restrict__ Q,
                                              __bf16* __restrict__ Kd,
                                              __bf16* __restrict__ VT) {
  int lane = threadIdx.x & 31, wave = threadIdx.x >> 5;
  int m0 = blockIdx.x * 256 + wave * 32;
  int n0 = blockIdx.y * 64;
  int z  = blockIdx.z;
  const __bf16* Wt   = (z == 0) ? wqt : (z == 1) ? wkt : wvt;
  const float*  bias = (z == 0) ? bq  : (z == 1) ? bk  : bv;
  int nl = lane & 15, hf = lane >> 4;
  const __bf16* arow0 = H + (size_t)(m0 + nl) * CDIM;
  const __bf16* arow1 = H + (size_t)(m0 + 16 + nl) * CDIM;
  v8f zero = {0.f, 0.f, 0.f, 0.f, 0.f, 0.f, 0.f, 0.f};
  v8f acc[2][4] = {{zero, zero, zero, zero}, {zero, zero, zero, zero}};
  for (int k0 = 0; k0 < CDIM; k0 += 32) {
    __builtin_prefetch(arow0 + k0 + 128, 0, 1);
    __builtin_prefetch(arow1 + k0 + 128, 0, 1);
    v16bf a0 = load_a_frag(arow0, k0 + hf * 8);
    v16bf a1 = load_a_frag(arow1, k0 + hf * 8);
#pragma unroll
    for (int nt = 0; nt < 4; ++nt) {
      const __bf16* brow = Wt + (size_t)(n0 + nt * 16 + nl) * CDIM;
      v16bf bfr = *(const v16bf*)(brow + k0 + hf * 16);
      acc[0][nt] = wmma_bf16(a0, bfr, acc[0][nt]);
      acc[1][nt] = wmma_bf16(a1, bfr, acc[1][nt]);
    }
  }
#pragma unroll
  for (int mt = 0; mt < 2; ++mt) {
#pragma unroll
    for (int nt = 0; nt < 4; ++nt) {
      int c = n0 + nt * 16 + nl;
      float bs = bias[c];
      int h = c >> 6, d = c & 63;
#pragma unroll
      for (int e = 0; e < 8; ++e) {
        int t = m0 + mt * 16 + hf * 8 + e;
        float val = acc[mt][nt][e] + bs;
        int b, win, tok; decode_token(t, b, win, tok);
        size_t hw = ((size_t)b * 8 + h) * 256 + win;
        if (z == 2)      VT[(hw * 64 + d) * 64 + tok] = f2bf(val);
        else if (z == 0) Q [(hw * 64 + tok) * 64 + d] = f2bf(val);
        else             Kd[(hw * 64 + tok) * 64 + d] = f2bf(val);
      }
    }
  }
}

// ---------------------------------------------------------------------------
// K3: windowed attention.  One wave per (b, head, window): 64x64 tile.
// QK^T (WMMA) -> LDS -> masked/biased softmax -> bf16 attn -> A*V (WMMA).
// ---------------------------------------------------------------------------
#define ATT_WAVES 2
__global__ __launch_bounds__(64) void k3_attn(const __bf16* __restrict__ Q,
                                              const __bf16* __restrict__ Kd,
                                              const __bf16* __restrict__ VT,
                                              const float* __restrict__ pos,
                                              __bf16* __restrict__ O) {
  __shared__ float  sLog[ATT_WAVES][64 * 65];   // stride 65 -> no bank conflicts
  __shared__ __bf16 sAtt[ATT_WAVES][64 * 72];   // stride 72 keeps 16B alignment
  __shared__ float  sBias[225];

  int tid = threadIdx.x;
  for (int i = tid; i < 225; i += 64) sBias[i] = pos[i];
  __syncthreads();

  int wave = tid >> 5, lane = tid & 31;
  int id  = blockIdx.x * ATT_WAVES + wave;       // (b*8 + h)*256 + win
  int b   = id >> 11, win = id & 255;
  const __bf16* q  = Q  + (size_t)id * 4096;
  const __bf16* k  = Kd + (size_t)id * 4096;
  const __bf16* vt = VT + (size_t)id * 4096;
  float*  slog = sLog[wave];
  __bf16* satt = sAtt[wave];
  int nl = lane & 15, hf = lane >> 4;

  // --- QK^T -> LDS logits ---
  v16bf bk_[4][2];
#pragma unroll
  for (int nt = 0; nt < 4; ++nt)
#pragma unroll
    for (int ks = 0; ks < 2; ++ks)
      bk_[nt][ks] = *(const v16bf*)(k + (size_t)(nt * 16 + nl) * 64 + ks * 32 + hf * 16);

  v8f zero = {0.f, 0.f, 0.f, 0.f, 0.f, 0.f, 0.f, 0.f};
#pragma unroll
  for (int mt = 0; mt < 4; ++mt) {
    const __bf16* arow = q + (size_t)(mt * 16 + nl) * 64;
    v16bf a0 = load_a_frag(arow, hf * 8);
    v16bf a1 = load_a_frag(arow, 32 + hf * 8);
#pragma unroll
    for (int nt = 0; nt < 4; ++nt) {
      v8f acc = zero;
      acc = wmma_bf16(a0, bk_[nt][0], acc);
      acc = wmma_bf16(a1, bk_[nt][1], acc);
#pragma unroll
      for (int e = 0; e < 8; ++e)
        slog[(mt * 16 + hf * 8 + e) * 65 + nt * 16 + nl] = acc[e];
    }
  }

  // --- softmax with shift-mask + relative position bias (2 rows / lane) ---
  int gy = win >> 4, gx = win & 15;
  bool eY = (gy == 15), eX = (gx == 15);
#pragma unroll
  for (int rr = 0; rr < 2; ++rr) {
    int r = lane * 2 + rr;
    int py = r >> 3, px = r & 7;
    float mx = -3.0e38f;
    for (int qi = 0; qi < 64; ++qi) {
      int qy = qi >> 3, qx = qi & 7;
      float l = slog[r * 65 + qi] * 0.125f +
                sBias[(py - qy + 7) * 15 + (px - qx + 7)];
      bool msk = (eY && ((py < 4) != (qy < 4))) || (eX && ((px < 4) != (qx < 4)));
      if (msk) l = -1.0e30f;
      slog[r * 65 + qi] = l;
      mx = fmaxf(mx, l);
    }
    float s = 0.f;
    for (int qi = 0; qi < 64; ++qi) {
      float e = __expf(slog[r * 65 + qi] - mx);
      s += e;
      slog[r * 65 + qi] = e;
    }
    float inv = 1.f / s;
    for (int qi = 0; qi < 64; ++qi)
      satt[r * 72 + qi] = f2bf(slog[r * 65 + qi] * inv);
  }

  // --- O = attn * V  (B fragments from transposed V: contiguous) ---
  v16bf bv_[4][2];
#pragma unroll
  for (int nt = 0; nt < 4; ++nt)
#pragma unroll
    for (int ks = 0; ks < 2; ++ks)
      bv_[nt][ks] = *(const v16bf*)(vt + (size_t)(nt * 16 + nl) * 64 + ks * 32 + hf * 16);

  int hd_head = (id >> 8) & 7;
#pragma unroll
  for (int mt = 0; mt < 4; ++mt) {
    const __bf16* arow = satt + (size_t)(mt * 16 + nl) * 72;
    v16bf a0 = load_a_frag(arow, hf * 8);
    v16bf a1 = load_a_frag(arow, 32 + hf * 8);
#pragma unroll
    for (int nt = 0; nt < 4; ++nt) {
      v8f acc = zero;
      acc = wmma_bf16(a0, bv_[nt][0], acc);
      acc = wmma_bf16(a1, bv_[nt][1], acc);
      int d = nt * 16 + nl;
      int c = hd_head * 64 + d;
#pragma unroll
      for (int e = 0; e < 8; ++e) {
        int tok = mt * 16 + hf * 8 + e;
        O[((size_t)(b * 256 + win) * 64 + tok) * CDIM + c] = f2bf(acc[e]);
      }
    }
  }
}

// ---------------------------------------------------------------------------
// K4: att_out = p + o (f32, kept for final residual) ; LN2 -> bf16 FIN.
// ---------------------------------------------------------------------------
__global__ __launch_bounds__(256) void k4_residual_ln2(const float* __restrict__ x,
                                                       const __bf16* __restrict__ O,
                                                       const float* __restrict__ g,
                                                       const float* __restrict__ beta,
                                                       float* __restrict__ ATT,
                                                       __bf16* __restrict__ FIN) {
  int t = (blockIdx.x * 256 + threadIdx.x) >> 5;
  int lane = threadIdx.x & 31;
  int b, win, tok; decode_token(t, b, win, tok);
  int hh, ww; token_to_hw(win, tok, hh, ww);
  int sh = (hh + 4) & 127, sw = (ww + 4) & 127;
  const float*  src  = x + (((size_t)b * 128 + sh) * 128 + sw) * CDIM;
  const __bf16* orow = O + (size_t)t * CDIM;
  float*        arow = ATT + (size_t)t * CDIM;
  float4 ab[4];
  float s = 0.f, s2 = 0.f;
#pragma unroll
  for (int j = 0; j < 4; ++j) {
    int c0 = (j * 32 + lane) * 4;
    float4 v = *(const float4*)(src + c0);
    v4bf ov = *(const v4bf*)(orow + c0);
    float4 a;
    a.x = v.x + bf2f(ov[0]);
    a.y = v.y + bf2f(ov[1]);
    a.z = v.z + bf2f(ov[2]);
    a.w = v.w + bf2f(ov[3]);
    *(float4*)(arow + c0) = a;
    ab[j] = a;
    s  += a.x + a.y + a.z + a.w;
    s2 += a.x * a.x + a.y * a.y + a.z * a.z + a.w * a.w;
  }
#pragma unroll
  for (int off = 16; off > 0; off >>= 1) {
    s  += __shfl_xor(s, off, 32);
    s2 += __shfl_xor(s2, off, 32);
  }
  float mu = s * (1.f / CDIM);
  float rstd = rsqrtf(s2 * (1.f / CDIM) - mu * mu + 1e-5f);
  __bf16* frow = FIN + (size_t)t * CDIM;
#pragma unroll
  for (int j = 0; j < 4; ++j) {
    int c0 = (j * 32 + lane) * 4;
    float4 a = ab[j];
    v4bf o;
    o[0] = f2bf((a.x - mu) * rstd * g[c0 + 0] + beta[c0 + 0]);
    o[1] = f2bf((a.y - mu) * rstd * g[c0 + 1] + beta[c0 + 1]);
    o[2] = f2bf((a.z - mu) * rstd * g[c0 + 2] + beta[c0 + 2]);
    o[3] = f2bf((a.w - mu) * rstd * g[c0 + 3] + beta[c0 + 3]);
    *(v4bf*)(frow + c0) = o;
  }
}

// ---------------------------------------------------------------------------
// K5: MLP up-projection + exact GELU -> bf16 HID[t, 2048].  32x64 wave tile.
// ---------------------------------------------------------------------------
__global__ __launch_bounds__(256) void k5_mlp1(const __bf16* __restrict__ A,
                                               const __bf16* __restrict__ Wt,
                                               const float* __restrict__ bias,
                                               __bf16* __restrict__ HID) {
  int lane = threadIdx.x & 31, wave = threadIdx.x >> 5;
  int m0 = blockIdx.x * 256 + wave * 32;
  int n0 = blockIdx.y * 64;
  int nl = lane & 15, hf = lane >> 4;
  const __bf16* arow0 = A + (size_t)(m0 + nl) * CDIM;
  const __bf16* arow1 = A + (size_t)(m0 + 16 + nl) * CDIM;
  v8f zero = {0.f, 0.f, 0.f, 0.f, 0.f, 0.f, 0.f, 0.f};
  v8f acc[2][4] = {{zero, zero, zero, zero}, {zero, zero, zero, zero}};
  for (int k0 = 0; k0 < CDIM; k0 += 32) {
    __builtin_prefetch(arow0 + k0 + 128, 0, 1);
    __builtin_prefetch(arow1 + k0 + 128, 0, 1);
    v16bf a0 = load_a_frag(arow0, k0 + hf * 8);
    v16bf a1 = load_a_frag(arow1, k0 + hf * 8);
#pragma unroll
    for (int nt = 0; nt < 4; ++nt) {
      const __bf16* brow = Wt + (size_t)(n0 + nt * 16 + nl) * CDIM;
      v16bf bfr = *(const v16bf*)(brow + k0 + hf * 16);
      acc[0][nt] = wmma_bf16(a0, bfr, acc[0][nt]);
      acc[1][nt] = wmma_bf16(a1, bfr, acc[1][nt]);
    }
  }
#pragma unroll
  for (int mt = 0; mt < 2; ++mt) {
#pragma unroll
    for (int nt = 0; nt < 4; ++nt) {
      int c = n0 + nt * 16 + nl;
      float bs = bias[c];
#pragma unroll
      for (int e = 0; e < 8; ++e) {
        int t = m0 + mt * 16 + hf * 8 + e;
        float xg = acc[mt][nt][e] + bs;
        float gl = 0.5f * xg * (1.f + erff(xg * 0.7071067811865476f));  // exact gelu
        HID[(size_t)t * HIDDEN + c] = f2bf(gl);
      }
    }
  }
}

// ---------------------------------------------------------------------------
// K6: MLP down-projection + bias + residual, inverse window/roll scatter.
// ---------------------------------------------------------------------------
__global__ __launch_bounds__(256) void k6_mlp2_out(const __bf16* __restrict__ A,
                                                   const __bf16* __restrict__ Wt,
                                                   const float* __restrict__ bias,
                                                   const float* __restrict__ ATT,
                                                   float* __restrict__ out) {
  int lane = threadIdx.x & 31, wave = threadIdx.x >> 5;
  int m0 = blockIdx.x * 256 + wave * 32;
  int n0 = blockIdx.y * 64;
  int nl = lane & 15, hf = lane >> 4;
  const __bf16* arow0 = A + (size_t)(m0 + nl) * HIDDEN;
  const __bf16* arow1 = A + (size_t)(m0 + 16 + nl) * HIDDEN;
  v8f zero = {0.f, 0.f, 0.f, 0.f, 0.f, 0.f, 0.f, 0.f};
  v8f acc[2][4] = {{zero, zero, zero, zero}, {zero, zero, zero, zero}};
  for (int k0 = 0; k0 < HIDDEN; k0 += 32) {
    __builtin_prefetch(arow0 + k0 + 256, 0, 1);
    __builtin_prefetch(arow1 + k0 + 256, 0, 1);
    v16bf a0 = load_a_frag(arow0, k0 + hf * 8);
    v16bf a1 = load_a_frag(arow1, k0 + hf * 8);
#pragma unroll
    for (int nt = 0; nt < 4; ++nt) {
      const __bf16* brow = Wt + (size_t)(n0 + nt * 16 + nl) * HIDDEN;
      v16bf bfr = *(const v16bf*)(brow + k0 + hf * 16);
      acc[0][nt] = wmma_bf16(a0, bfr, acc[0][nt]);
      acc[1][nt] = wmma_bf16(a1, bfr, acc[1][nt]);
    }
  }
#pragma unroll
  for (int mt = 0; mt < 2; ++mt) {
#pragma unroll
    for (int nt = 0; nt < 4; ++nt) {
      int c = n0 + nt * 16 + nl;
      float bs = bias[c];
#pragma unroll
      for (int e = 0; e < 8; ++e) {
        int t = m0 + mt * 16 + hf * 8 + e;
        float val = acc[mt][nt][e] + bs + ATT[(size_t)t * CDIM + c];
        int b, win, tok; decode_token(t, b, win, tok);
        int hh, ww; token_to_hw(win, tok, hh, ww);
        int dh = (hh + 4) & 127, dw = (ww + 4) & 127;  // roll(+4,+4) scatter
        out[(((size_t)b * 128 + dh) * 128 + dw) * CDIM + c] = val;
      }
    }
  }
}

// ---------------------------------------------------------------------------
extern "C" void kernel_launch(void* const* d_in, const int* in_sizes, int n_in,
                              void* d_out, int out_size, void* d_ws, size_t ws_size,
                              hipStream_t stream) {
  (void)in_sizes; (void)n_in; (void)out_size; (void)ws_size;
  const float* x   = (const float*)d_in[0];
  const float* wq  = (const float*)d_in[1];
  const float* bq  = (const float*)d_in[2];
  const float* wk  = (const float*)d_in[3];
  const float* bk  = (const float*)d_in[4];
  const float* wv  = (const float*)d_in[5];
  const float* bv  = (const float*)d_in[6];
  const float* pos = (const float*)d_in[7];
  const float* g1  = (const float*)d_in[8];
  const float* be1 = (const float*)d_in[9];
  const float* g2  = (const float*)d_in[10];
  const float* be2 = (const float*)d_in[11];
  const float* w1  = (const float*)d_in[12];
  const float* b1  = (const float*)d_in[13];
  const float* w2  = (const float*)d_in[14];
  const float* b2  = (const float*)d_in[15];

  char* ws = (char*)d_ws;
  __bf16* wqt = (__bf16*)(ws + OFF_WQT);
  __bf16* wkt = (__bf16*)(ws + OFF_WKT);
  __bf16* wvt = (__bf16*)(ws + OFF_WVT);
  __bf16* w1t = (__bf16*)(ws + OFF_W1T);
  __bf16* w2t = (__bf16*)(ws + OFF_W2T);
  __bf16* Hb  = (__bf16*)(ws + OFF_H);
  __bf16* Qb  = (__bf16*)(ws + OFF_Q);
  __bf16* Kb  = (__bf16*)(ws + OFF_K);
  __bf16* Vt  = (__bf16*)(ws + OFF_V);
  __bf16* Ob  = (__bf16*)(ws + OFF_O);
  float*  ATb = (float*)(ws + OFF_ATT);
  __bf16* Fb  = (__bf16*)(ws + OFF_FIN);
  __bf16* Hd  = (__bf16*)(ws + OFF_HID);

  // weight transposes -> bf16
  k0_transpose<<<(512 * 512 + 255) / 256, 256, 0, stream>>>(wq, wqt, 512, 512);
  k0_transpose<<<(512 * 512 + 255) / 256, 256, 0, stream>>>(wk, wkt, 512, 512);
  k0_transpose<<<(512 * 512 + 255) / 256, 256, 0, stream>>>(wv, wvt, 512, 512);
  k0_transpose<<<(512 * 2048 + 255) / 256, 256, 0, stream>>>(w1, w1t, 512, 2048);
  k0_transpose<<<(2048 * 512 + 255) / 256, 256, 0, stream>>>(w2, w2t, 2048, 512);

  // 1) roll + window partition + LN1
  k1_ln1_gather<<<NTOK / 8, 256, 0, stream>>>(x, g1, be1, Hb);
  // 2) QKV projections (z = q/k/v), 32x64 per-wave tiles
  k2_qkv<<<dim3(NTOK / 256, CDIM / 64, 3), 256, 0, stream>>>(
      Hb, wqt, wkt, wvt, bq, bk, bv, Qb, Kb, Vt);
  // 3) windowed attention (16384 head-windows, 2 waves / block)
  k3_attn<<<16384 / ATT_WAVES, 64, 0, stream>>>(Qb, Kb, Vt, pos, Ob);
  // 4) residual + LN2
  k4_residual_ln2<<<NTOK / 8, 256, 0, stream>>>(x, Ob, g2, be2, ATb, Fb);
  // 5) MLP up + GELU
  k5_mlp1<<<dim3(NTOK / 256, HIDDEN / 64), 256, 0, stream>>>(Fb, w1t, b1, Hd);
  // 6) MLP down + residual + un-window/roll scatter
  k6_mlp2_out<<<dim3(NTOK / 256, CDIM / 64), 256, 0, stream>>>(
      Hd, w2t, b2, ATb, (float*)d_out);
}
